// Spatial_Attention_layer_71725953843986
// MI455X (gfx1250) — matrix operand order repaired and checked by reading, not compiled, not run on hardware
//
#include <hip/hip_runtime.h>
#include <hip/hip_bf16.h>
#include <math.h>

// ---------------------------------------------------------------------------
// Spatial attention:  B=32, N=512, F=64, T=24
//   lhs[b,n,t] = sum_f (sum_t' x[b,n,f,t']*W1[t']) * W2[f,t]
//   rhs[b,n,t] = sum_f W3[f] * x[b,n,f,t]
//   P[b,n,m]   = sum_t lhs[b,n,t]*rhs[b,m,t]
//   out        = softmax_n( sigmoid(P + bs) )
// Memory-bound (~137MB min traffic, ~0.55 GFLOP) -> stream x once, WMMA GEMM
// from tiny cached f16 factors, fused sigmoid/exp epilogue (v_rcp_f32 instead
// of IEEE divide), atomic column sums + rescale for the softmax over n.
// ---------------------------------------------------------------------------

#define Bsz 32
#define Nsz 512
#define Fsz 64
#define Tsz 24
#define TP  32   // padded K (f16), zeros in [24,32)

typedef _Float16 v8h  __attribute__((ext_vector_type(8)));
typedef _Float16 v16h __attribute__((ext_vector_type(16)));
typedef float    v8f  __attribute__((ext_vector_type(8)));

// ---------------------------------------------------------------------------
// Phase 0: zero the softmax column-sum accumulator (B*N floats)
// ---------------------------------------------------------------------------
__global__ void sa_zero_colsum(float* colsum) {
    colsum[blockIdx.x * blockDim.x + threadIdx.x] = 0.0f;
}

// ---------------------------------------------------------------------------
// Phase 1: stream x (100.7 MB) once; one wave32 per (b,n) pair.
// Produces f16 lhs/rhs rows of 32 halves (zero padded K=24..31).
// ---------------------------------------------------------------------------
__global__ __launch_bounds__(256) void sa_phase1(
    const float* __restrict__ x,  const float* __restrict__ W1,
    const float* __restrict__ W2, const float* __restrict__ W3,
    _Float16* __restrict__ lhs16, _Float16* __restrict__ rhs16) {

    __shared__ __align__(16) float s_w2[Fsz * Tsz];   // 6 KB, block-shared
    __shared__ float s_w1[Tsz];
    __shared__ float s_w3[Fsz];
    __shared__ __align__(16) float s_tile[8][Fsz * Tsz]; // 8 waves x 6 KB
    __shared__ float s_a[8][Fsz];

    const int tid  = threadIdx.x;
    const int lane = tid & 31;      // gfx1250 is wave32
    const int w    = tid >> 5;      // wave in block (0..7)

    // Cooperative load of the small weights into LDS.
    for (int idx = tid; idx < Fsz * Tsz; idx += 256) s_w2[idx] = W2[idx];
    if (tid < Tsz)                  s_w1[tid]      = W1[tid];
    if (tid >= 32 && tid < 32+Fsz)  s_w3[tid - 32] = W3[tid - 32];

    // Per-wave (b,n) pair; tile = 64x24 floats = 1536 contiguous floats.
    const int p = blockIdx.x * 8 + w;
    const float* xt = x + (size_t)p * (Fsz * Tsz);

#pragma unroll
    for (int k = 0; k < 12; ++k) {  // 12 * 32 lanes * float4 = 1536
        float4 v = *(const float4*)(xt + k * 128 + lane * 4);
        *(float4*)(&s_tile[w][k * 128 + lane * 4]) = v;
    }
    __syncthreads();

    // a[f] = dot(x[f,:], W1); lane handles f=lane and f=lane+32.
    {
        float a0 = 0.f, a1 = 0.f;
        const float* t0 = &s_tile[w][lane * Tsz];
        const float* t1 = &s_tile[w][(lane + 32) * Tsz];
#pragma unroll
        for (int t = 0; t < Tsz; ++t) {
            const float wv = s_w1[t];
            a0 += t0[t] * wv;
            a1 += t1[t] * wv;
        }
        s_a[w][lane]      = a0;
        s_a[w][lane + 32] = a1;
    }
    __syncthreads();

    // lanes t<24: lhs[t] = sum_f a[f]*W2[f,t]; rhs[t] = sum_f x[f,t]*W3[f]
    _Float16 hl = (_Float16)0.f, hr = (_Float16)0.f;
    if (lane < Tsz) {
        float l = 0.f, r = 0.f;
#pragma unroll
        for (int f = 0; f < Fsz; ++f) {
            const float av = s_a[w][f];
            l += av * s_w2[f * Tsz + lane];
            r += s_tile[w][f * Tsz + lane] * s_w3[f];
        }
        hl = (_Float16)l;
        hr = (_Float16)r;
    }
    // lanes 24..31 store the K padding zeros.
    lhs16[(size_t)p * TP + lane] = hl;
    rhs16[(size_t)p * TP + lane] = hr;
}

// ---------------------------------------------------------------------------
// Phase 2: batched (512x24)x(24x512) GEMM via V_WMMA_F32_16X16X32_F16
// (K padded 24->32), bs preloaded into the accumulator, fused
// sigmoid -> exp epilogue, per-column partial sums via f32 atomics.
// One wave per (b, row-tile i); loops over 32 column tiles j.
// ---------------------------------------------------------------------------
__global__ __launch_bounds__(256) void sa_phase2(
    const _Float16* __restrict__ lhs16, const _Float16* __restrict__ rhs16,
    const float* __restrict__ bs, float* __restrict__ out,
    float* __restrict__ colsum) {

    const int tid     = threadIdx.x;
    const int lane    = tid & 31;
    const int waveId  = blockIdx.x * (blockDim.x >> 5) + (tid >> 5);
    const int b       = waveId >> 5;   // 0..31
    const int i       = waveId & 31;   // row tile 0..31
    const int halfSel = lane >> 4;     // 0: lanes 0-15, 1: lanes 16-31
    const int lr      = lane & 15;

    // --- A fragment (lhs rows 16i..16i+15), ISA 16-bit A 16x32 layout ---
    // lane<16 : v0..3 = K0..7, v4..7 = K16..23 ; lane>=16: K8..15, K24..31(=0)
    const _Float16* aRow = lhs16 + (size_t)(b * Nsz + i * 16 + lr) * TP;
    const int aOff0 = halfSel ? 8  : 0;
    const int aOff1 = halfSel ? 24 : 16;
    v8h alo = *(const v8h*)(aRow + aOff0);
    v8h ahi = *(const v8h*)(aRow + aOff1);
    v16h A;
#pragma unroll
    for (int k = 0; k < 8; ++k) { A[k] = alo[k]; A[k + 8] = ahi[k]; }

    // --- B fragment offsets (rhs rows = columns m), ISA 32x16 B layout ---
    // lane<16 : K0..7, K16..23 (half-offsets 0,16) ; lane>=16 : K8..15, K24..31 (8,24)
    const int bOff0 = halfSel ? 8  : 0;
    const int bOff1 = halfSel ? 24 : 16;

    float* outB = out + (size_t)b * (Nsz * Nsz);
    const int row0 = i * 16 + halfSel * 8;   // C/D layout: M = vgpr + 8*halfSel

    for (int j = 0; j < 32; ++j) {
        const _Float16* bRow = rhs16 + (size_t)(b * Nsz + j * 16 + lr) * TP;
        if (j < 31) __builtin_prefetch(bRow + 16 * TP, 0, 1); // global_prefetch_b8
        v8h blo = *(const v8h*)(bRow + bOff0);
        v8h bhi = *(const v8h*)(bRow + bOff1);
        v16h Bf;
#pragma unroll
        for (int k = 0; k < 8; ++k) { Bf[k] = blo[k]; Bf[k + 8] = bhi[k]; }

        const int col = j * 16 + lr;         // C/D layout: N = lane&15

        // Preload bias into the accumulator: product + bs is free.
        v8f C;
#pragma unroll
        for (int r = 0; r < 8; ++r) C[r] = bs[(size_t)(row0 + r) * Nsz + col];

        C = __builtin_amdgcn_wmma_f32_16x16x32_f16(
                /*neg_a=*/false, A, /*neg_b=*/false, Bf,
                /*c_mod=*/(short)0, C, /*reuse_a=*/false, /*reuse_b=*/false);

        // Fused epilogue: e = exp(sigmoid(P+bs)); softmax numerator.
        // v_rcp_f32 (fast, ~1ulp) instead of the IEEE divide expansion.
        float psum = 0.f;
#pragma unroll
        for (int r = 0; r < 8; ++r) {
            const float s = __builtin_amdgcn_rcpf(1.0f + __expf(-C[r]));
            const float e = __expf(s);
            psum += e;
            outB[(size_t)(row0 + r) * Nsz + col] = e;
        }
        atomicAdd(&colsum[b * Nsz + col], psum);  // global_atomic_add_f32
    }
}

// ---------------------------------------------------------------------------
// Phase 3: softmax denominator — out[b,n,m] *= rcp(colsum[b,m])
// ---------------------------------------------------------------------------
__global__ void sa_phase3(float* __restrict__ out,
                          const float* __restrict__ colsum) {
    const int idx = blockIdx.x * blockDim.x + threadIdx.x;
    const int m = idx & (Nsz - 1);
    const int b = idx >> 18;                  // / (512*512)
    out[idx] = out[idx] * __builtin_amdgcn_rcpf(colsum[b * Nsz + m]);
}

// ---------------------------------------------------------------------------
extern "C" void kernel_launch(void* const* d_in, const int* in_sizes, int n_in,
                              void* d_out, int out_size, void* d_ws, size_t ws_size,
                              hipStream_t stream) {
    const float* x  = (const float*)d_in[0];  // (32,512,64,24)
    const float* W1 = (const float*)d_in[1];  // (24,)
    const float* W2 = (const float*)d_in[2];  // (64,24)
    const float* W3 = (const float*)d_in[3];  // (64,)
    const float* bs = (const float*)d_in[4];  // (1,512,512)
    float* out = (float*)d_out;               // (32,512,512)

    char* ws = (char*)d_ws;
    _Float16* lhs16 = (_Float16*)(ws);                     // 16384*32*2 = 1 MB
    _Float16* rhs16 = (_Float16*)(ws + (1u << 20));        // 1 MB
    float*    colsum = (float*)(ws + (2u << 20));          // 32*512*4 = 64 KB

    sa_zero_colsum<<<(Bsz * Nsz) / 256, 256, 0, stream>>>(colsum);
    sa_phase1<<<(Bsz * Nsz) / 8, 256, 0, stream>>>(x, W1, W2, W3, lhs16, rhs16);
    sa_phase2<<<(Bsz * 32) / 8, 256, 0, stream>>>(lhs16, rhs16, bs, out, colsum);
    sa_phase3<<<(Bsz * Nsz * Nsz) / 256, 256, 0, stream>>>(out, colsum);
}
